// MSARowAttentionWithPairBiasOpt_77043123355720
// MI455X (gfx1250) — matrix operand order repaired and checked by reading, not compiled, not run on hardware
//
#include <hip/hip_runtime.h>

#define NSEQ 128
#define NRES 256
#define MSA_DIM 256
#define PAIR_DIM 128
#define NHEAD 8
#define HEAD_DIM 32

typedef _Float16 half_t;
typedef __attribute__((ext_vector_type(16))) _Float16 v16h;
typedef __attribute__((ext_vector_type(8)))  _Float16 v8h;
typedef __attribute__((ext_vector_type(8)))  float    v8f;

#define CAT16(lo, hi) __builtin_shufflevector((lo), (hi), 0,1,2,3,4,5,6,7,8,9,10,11,12,13,14,15)

// ---------------------------------------------------------------- utilities

// Transpose-convert a 256x256 f32 matrix to f16: d[n*256+k] = s[k*256+n].
__global__ void k_w_to_f16t(const float* __restrict__ s, half_t* __restrict__ d) {
  int i = blockIdx.x * 256 + threadIdx.x;           // 65536 threads
  int n = i >> 8, k = i & 255;
  d[i] = (half_t)s[(size_t)k * 256 + n];
}

// LayerNorm over last dim (256) of msa_act, emit f16. One wave32 per row.
__global__ void k_ln_msa(const float* __restrict__ x, const float* __restrict__ sc,
                         const float* __restrict__ bi, half_t* __restrict__ out) {
  int lane = threadIdx.x & 31;
  int row  = blockIdx.x * 8 + (threadIdx.x >> 5);
  const float* p = x + (size_t)row * MSA_DIM;
  float v[8];
  float s = 0.f, s2 = 0.f;
#pragma unroll
  for (int j = 0; j < 8; ++j) { float t = p[lane * 8 + j]; v[j] = t; s += t; s2 += t * t; }
#pragma unroll
  for (int o = 16; o > 0; o >>= 1) { s += __shfl_xor(s, o, 32); s2 += __shfl_xor(s2, o, 32); }
  float mu  = s  * (1.f / MSA_DIM);
  float var = s2 * (1.f / MSA_DIM) - mu * mu;
  float inv = rsqrtf(var + 1e-5f);
#pragma unroll
  for (int j = 0; j < 8; ++j) {
    int c = lane * 8 + j;
    out[(size_t)row * MSA_DIM + c] = (half_t)((v[j] - mu) * inv * sc[c] + bi[c]);
  }
}

// Pair LayerNorm + einsum('qkc,ch->hqk'). One wave32 per (q,k) position.
__global__ void k_pair_bias(const float* __restrict__ pair, const float* __restrict__ sc,
                            const float* __restrict__ bi, const float* __restrict__ w,
                            float* __restrict__ nb) {
  int lane = threadIdx.x & 31;
  int idx  = blockIdx.x * 8 + (threadIdx.x >> 5);   // q*256 + k
  const float* p = pair + (size_t)idx * PAIR_DIM;
  float v[4];
  float s = 0.f, s2 = 0.f;
#pragma unroll
  for (int j = 0; j < 4; ++j) { float t = p[lane * 4 + j]; v[j] = t; s += t; s2 += t * t; }
#pragma unroll
  for (int o = 16; o > 0; o >>= 1) { s += __shfl_xor(s, o, 32); s2 += __shfl_xor(s2, o, 32); }
  float mu  = s  * (1.f / PAIR_DIM);
  float var = s2 * (1.f / PAIR_DIM) - mu * mu;
  float inv = rsqrtf(var + 1e-5f);
  float pn[4];
#pragma unroll
  for (int j = 0; j < 4; ++j) {
    int c = lane * 4 + j;
    pn[j] = (v[j] - mu) * inv * sc[c] + bi[c];
  }
#pragma unroll
  for (int h = 0; h < NHEAD; ++h) {
    float a = 0.f;
#pragma unroll
    for (int j = 0; j < 4; ++j) a += pn[j] * w[(lane * 4 + j) * NHEAD + h];
#pragma unroll
    for (int o = 16; o > 0; o >>= 1) a += __shfl_xor(a, o, 32);
    if (lane == 0) nb[(size_t)h * NRES * NRES + idx] = a;
  }
}

// ------------------------------------------------------------- WMMA helpers

// A fragment of a 16x32 f16 tile from a row-major matrix (two b128 loads).
__device__ inline v16h load_a_frag(const half_t* __restrict__ base, int row, int ldm, int kb) {
  v8h lo = *(const v8h*)(base + (size_t)row * ldm + kb);
  v8h hi = *(const v8h*)(base + (size_t)row * ldm + kb + 8);
  return CAT16(lo, hi);
}

// -------------------------------------------------------- projection GEMMs
// X: [32768 x 256] f16 row-major. WT: [256 x 256] f16, TRANSPOSED (WT[n][k]).
// kind: 0 = Q (scale), 1 = K (plain), 2 = gate (sigmoid(x+gb)), 3 = V
// (store head-transposed vT[b][h][c][key]).  Each wave: 16 rows x 4 n-tiles.
__global__ void k_proj_gemm(const half_t* __restrict__ X, const half_t* __restrict__ WT,
                            const float* __restrict__ gb, half_t* __restrict__ out, int kind) {
  int lane = threadIdx.x & 31;
  int tm  = blockIdx.x;        // 0..2047
  int tn4 = blockIdx.y;        // 0..3  -> n-tiles tn4*4 .. tn4*4+3
  int mrow = tm * 16 + (lane & 15);
  v8f acc[4] = {{}, {}, {}, {}};
#pragma unroll
  for (int kk = 0; kk < MSA_DIM / 32; ++kk) {
    int kbA = kk * 32 + ((lane & 16) ? 16 : 0);
    v16h a = load_a_frag(X, mrow, MSA_DIM, kbA);
    int kbB = kk * 32 + ((lane & 16) ? 16 : 0);
#pragma unroll
    for (int t = 0; t < 4; ++t) {
      int ncol = (tn4 * 4 + t) * 16 + (lane & 15);
      v16h b = *(const v16h*)(WT + (size_t)ncol * MSA_DIM + kbB);
      acc[t] = __builtin_amdgcn_wmma_f32_16x16x32_f16(false, a, false, b, (short)0, acc[t], false, false);
    }
  }
#pragma unroll
  for (int t = 0; t < 4; ++t) {
    int ncol = (tn4 * 4 + t) * 16 + (lane & 15);
    int h = ncol >> 5, c = ncol & 31;
#pragma unroll
    for (int e = 0; e < 8; ++e) {
      int row = tm * 16 + e + ((lane & 16) ? 8 : 0);
      int bb = row >> 8, q = row & 255;
      float val = acc[t][e];
      if (kind == 0)      val *= 0.17677669529663687f;          // 1/sqrt(32)
      else if (kind == 2) val = 1.f / (1.f + __expf(-(val + gb[ncol])));
      if (kind == 3)      // V: head-transposed [b][h][c][key]
        out[(((size_t)(bb * NHEAD + h)) * HEAD_DIM + c) * NRES + q] = (half_t)val;
      else                // Q/K/gate: [b][h][key][c]
        out[(((size_t)(bb * NHEAD + h)) * NRES + q) * HEAD_DIM + c] = (half_t)val;
    }
  }
}

// Output GEMM: ah [32768 x 256] f16 x o_w16T [256 x 256] + o_b -> f32 out.
__global__ void k_out_gemm(const half_t* __restrict__ X, const half_t* __restrict__ WT,
                           const float* __restrict__ ob, float* __restrict__ out) {
  int lane = threadIdx.x & 31;
  int tm  = blockIdx.x;
  int tn4 = blockIdx.y;
  int mrow = tm * 16 + (lane & 15);
  v8f acc[4] = {{}, {}, {}, {}};
#pragma unroll
  for (int kk = 0; kk < MSA_DIM / 32; ++kk) {
    int kbA = kk * 32 + ((lane & 16) ? 16 : 0);
    v16h a = load_a_frag(X, mrow, MSA_DIM, kbA);
    int kbB = kk * 32 + ((lane & 16) ? 16 : 0);
#pragma unroll
    for (int t = 0; t < 4; ++t) {
      int ncol = (tn4 * 4 + t) * 16 + (lane & 15);
      v16h b = *(const v16h*)(WT + (size_t)ncol * MSA_DIM + kbB);
      acc[t] = __builtin_amdgcn_wmma_f32_16x16x32_f16(false, a, false, b, (short)0, acc[t], false, false);
    }
  }
#pragma unroll
  for (int t = 0; t < 4; ++t) {
    int ncol = (tn4 * 4 + t) * 16 + (lane & 15);
#pragma unroll
    for (int e = 0; e < 8; ++e) {
      int row = tm * 16 + e + ((lane & 16) ? 8 : 0);
      out[(size_t)row * MSA_DIM + ncol] = acc[t][e] + ob[ncol];
    }
  }
}

// ------------------------------------------------------------- attention
// One wave per (qtile, head, batch): logits (16x256) via WMMA -> LDS,
// softmax, then PV via WMMA, gate in epilogue.
// kh: [b][h][key][c] f16.  vT: [b][h][c][key] f16 (head-transposed).
__global__ void k_attn(const half_t* __restrict__ qh, const half_t* __restrict__ kh,
                       const half_t* __restrict__ vT, const half_t* __restrict__ gh,
                       const float* __restrict__ nb, const float* __restrict__ mask,
                       half_t* __restrict__ ah) {
  __shared__ float  lg[16][NRES + 8];
  __shared__ half_t wt[16][NRES];
  int lane = threadIdx.x & 31;
  int qt = blockIdx.x, h = blockIdx.y, b = blockIdx.z;
  const size_t bh = (size_t)(b * NHEAD + h) * NRES;

  int mq  = qt * 16 + (lane & 15);
  int kbA = (lane & 16) ? 16 : 0;
  v16h aq = load_a_frag(qh + bh * HEAD_DIM, mq, HEAD_DIM, kbA);

  // logits = q . k^T + pair bias + mask bias
  for (int kt = 0; kt < 16; ++kt) {
    int key = kt * 16 + (lane & 15);
    int kbB = (lane & 16) ? 16 : 0;
    v16h bk = *(const v16h*)(kh + (bh + key) * HEAD_DIM + kbB);
    v8f acc = {};
    acc = __builtin_amdgcn_wmma_f32_16x16x32_f16(false, aq, false, bk, (short)0, acc, false, false);
#pragma unroll
    for (int e = 0; e < 8; ++e) {
      int rloc = e + ((lane & 16) ? 8 : 0);
      int qg = qt * 16 + rloc;
      int kg = kt * 16 + (lane & 15);
      lg[rloc][kg] = acc[e] + nb[((size_t)h * NRES + qg) * NRES + kg]
                   + 1e9f * (mask[(size_t)b * NRES + kg] - 1.0f);
    }
  }
  __syncthreads();

  // row softmax -> f16 weights in LDS (two lanes per row)
  {
    int r = lane & 15;
    int k0 = (lane >> 4) * 128;
    float mx = -3.4e38f;
    for (int k = k0; k < k0 + 128; ++k) mx = fmaxf(mx, lg[r][k]);
    mx = fmaxf(mx, __shfl_xor(mx, 16, 32));
    float s = 0.f;
    for (int k = k0; k < k0 + 128; ++k) { float e = __expf(lg[r][k] - mx); lg[r][k] = e; s += e; }
    s += __shfl_xor(s, 16, 32);
    float inv = 1.f / s;
    for (int k = k0; k < k0 + 128; ++k) wt[r][k] = (half_t)(lg[r][k] * inv);
  }
  __syncthreads();

  // weighted_avg = weights x V  (16 x 256) x (256 x 32), two 16-wide N tiles
  v8f av0 = {}, av1 = {};
  const half_t* vThead = vT + bh * HEAD_DIM;   // [c][key], c-major
  for (int kc = 0; kc < 8; ++kc) {
    int m  = lane & 15;
    int kb = kc * 32 + ((lane & 16) ? 16 : 0);
    v8h wlo = *(const v8h*)&wt[m][kb];
    v8h whi = *(const v8h*)&wt[m][kb + 8];
    v16h aw = CAT16(wlo, whi);
    int kb2 = kc * 32 + ((lane & 16) ? 16 : 0);
    int c0 = lane & 15;
    v16h bv0 = *(const v16h*)(vThead + (size_t)c0 * NRES + kb2);
    v16h bv1 = *(const v16h*)(vThead + (size_t)(16 + c0) * NRES + kb2);
    av0 = __builtin_amdgcn_wmma_f32_16x16x32_f16(false, aw, false, bv0, (short)0, av0, false, false);
    av1 = __builtin_amdgcn_wmma_f32_16x16x32_f16(false, aw, false, bv1, (short)0, av1, false, false);
  }

  // gate + store to ah [b][q][h*32+c] f16
#pragma unroll
  for (int e = 0; e < 8; ++e) {
    int qg = qt * 16 + e + ((lane & 16) ? 8 : 0);
    int c0 = lane & 15;
    float g0 = (float)gh[(bh + qg) * HEAD_DIM + c0];
    float g1 = (float)gh[(bh + qg) * HEAD_DIM + 16 + c0];
    size_t orow = ((size_t)(b * NRES + qg)) * MSA_DIM + h * HEAD_DIM;
    ah[orow + c0]      = (half_t)(av0[e] * g0);
    ah[orow + 16 + c0] = (half_t)(av1[e] * g1);
  }
}

// ---------------------------------------------------------------- launcher

extern "C" void kernel_launch(void* const* d_in, const int* in_sizes, int n_in,
                              void* d_out, int out_size, void* d_ws, size_t ws_size,
                              hipStream_t stream) {
  (void)in_sizes; (void)n_in; (void)out_size; (void)ws_size;
  const float* msa  = (const float*)d_in[0];
  const float* mask = (const float*)d_in[1];
  const float* pair = (const float*)d_in[2];
  const float* qns  = (const float*)d_in[3];
  const float* qnb  = (const float*)d_in[4];
  const float* pns  = (const float*)d_in[5];
  const float* pnb  = (const float*)d_in[6];
  const float* f2d  = (const float*)d_in[7];
  const float* qw   = (const float*)d_in[8];
  const float* kw   = (const float*)d_in[9];
  const float* vw   = (const float*)d_in[10];
  const float* gw   = (const float*)d_in[11];
  const float* gb   = (const float*)d_in[12];
  const float* ow   = (const float*)d_in[13];
  const float* ob   = (const float*)d_in[14];
  float* out = (float*)d_out;

  char* ws = (char*)d_ws;
  size_t off = 0;
  auto alloc = [&](size_t bytes) -> void* {
    void* p = ws + off;
    off += (bytes + 255) & ~(size_t)255;
    return p;
  };
  const size_t NROW = (size_t)NSEQ * NRES;                // 32768
  half_t* xh    = (half_t*)alloc(NROW * MSA_DIM * 2);     // LN(msa) f16
  half_t* wq16t = (half_t*)alloc(65536 * 2);              // transposed f16 weights
  half_t* wk16t = (half_t*)alloc(65536 * 2);
  half_t* wv16t = (half_t*)alloc(65536 * 2);
  half_t* wg16t = (half_t*)alloc(65536 * 2);
  half_t* wo16t = (half_t*)alloc(65536 * 2);
  half_t* qhb   = (half_t*)alloc(NROW * MSA_DIM * 2);     // [b][h][q][32]
  half_t* khb   = (half_t*)alloc(NROW * MSA_DIM * 2);     // [b][h][key][32]
  half_t* vTb   = (half_t*)alloc(NROW * MSA_DIM * 2);     // [b][h][c][key]
  half_t* ghb   = (half_t*)alloc(NROW * MSA_DIM * 2);     // [b][h][q][32]
  half_t* ahb   = (half_t*)alloc(NROW * MSA_DIM * 2);     // gated attn out, [b][q][256]
  float*  nbb   = (float*)alloc((size_t)NHEAD * NRES * NRES * 4);

  // 1) transposed f16 weight copies
  k_w_to_f16t<<<256, 256, 0, stream>>>(qw, wq16t);
  k_w_to_f16t<<<256, 256, 0, stream>>>(kw, wk16t);
  k_w_to_f16t<<<256, 256, 0, stream>>>(vw, wv16t);
  k_w_to_f16t<<<256, 256, 0, stream>>>(gw, wg16t);
  k_w_to_f16t<<<256, 256, 0, stream>>>(ow, wo16t);

  // 2) LayerNorms + pair bias
  k_ln_msa<<<NROW / 8, 256, 0, stream>>>(msa, qns, qnb, xh);
  k_pair_bias<<<(NRES * NRES) / 8, 256, 0, stream>>>(pair, pns, pnb, f2d, nbb);

  // 3) Q/K/V/gate projections (WMMA)
  dim3 pg(NROW / 16, 4);
  k_proj_gemm<<<pg, 32, 0, stream>>>(xh, wq16t, gb, qhb, 0);
  k_proj_gemm<<<pg, 32, 0, stream>>>(xh, wk16t, gb, khb, 1);
  k_proj_gemm<<<pg, 32, 0, stream>>>(xh, wv16t, gb, vTb, 3);
  k_proj_gemm<<<pg, 32, 0, stream>>>(xh, wg16t, gb, ghb, 2);

  // 4) attention (WMMA logits + softmax + WMMA PV + gate)
  dim3 ag(NRES / 16, NHEAD, NSEQ);
  k_attn<<<ag, 32, 0, stream>>>(qhb, khb, vTb, ghb, nbb, mask, ahb);

  // 5) output projection (WMMA) + o_b
  k_out_gemm<<<pg, 32, 0, stream>>>(ahb, wo16t, ob, out);
}
